// Ms2z_64622077935692
// MI455X (gfx1250) — matrix operand order repaired and checked by reading, not compiled, not run on hardware
//
#include <hip/hip_runtime.h>

// ---------------------------------------------------------------------------
// MI455X (gfx1250) implementation.
//
// word_emb = (emb[inp] + memory) @ Wout + bout = embW[inp] + mW[b]
//   embW = emb @ Wout  (6.55 GFLOP, the only compute-heavy op)
//   mW   = memory @ Wout + bout
// emb + embW = 102 MB -> L2-resident (192 MB); HBM traffic ~150 MB ~ 7 us.
// GEMMs use v_wmma_f32_16x16x32_f16 with a block-tiled LDS-staged B panel:
//   block = 256 thr (8 waves), tile 128x64, BK=32; B k-slab converted to f16
//   and transposed into 4 KB LDS; each wave: 4 n-tiles -> 4 WMMAs / K-step.
// ---------------------------------------------------------------------------

typedef __attribute__((ext_vector_type(16))) _Float16 v16h;
typedef __attribute__((ext_vector_type(8)))  _Float16 v8h;
typedef __attribute__((ext_vector_type(8)))  float    v8f;

#define PAD_ID 0
#define BOS_ID 1
#define EOS_ID 2
#define Bsz 256
#define Ssz 512
#define Vsz 50000
#define Dsz 256
#define Lsz 128

// ---------------------------------------------------------------------------
// enc[b,d] = sum_s mask*(emb[vt]+emb[parent]) / max(cnt,1); also ffalse[b].
// ---------------------------------------------------------------------------
__global__ __launch_bounds__(256) void enc_kernel(
    const int* __restrict__ vocab, const int* __restrict__ order,
    const int* __restrict__ mask, const float* __restrict__ emb,
    float* __restrict__ enc, int* __restrict__ flen) {
  int b = blockIdx.x;
  int d = threadIdx.x;
  const int* mrow = mask + b * Ssz;
  const int* vrow = vocab + b * Ssz;
  float acc = 0.0f;
  int cnt = 0;
  int ffalse = Ssz;
  for (int s = 0; s < Ssz; ++s) {
    int m = mrow[s];
    if (!m) {
      if (s < ffalse) ffalse = s;
      continue;
    }
    ++cnt;
    int vt = vrow[s];                         // mask true -> raw vocab
    int oi = order[(b * Ssz + s) * 6 + 0];    // parent index
    if (oi == -1) oi = BOS_ID;                // jnp.where(oi==-1, BOS, oi)
    int pv = mrow[oi] ? vrow[oi] : PAD_ID;    // vt[oi] (vt is mask-gated)
    int parent = (s == 0) ? BOS_ID : pv;      // parent[:,0] = BOS
    acc += emb[(size_t)vt * Dsz + d] + emb[(size_t)parent * Dsz + d];
  }
  enc[b * Dsz + d] = acc / (cnt > 0 ? (float)cnt : 1.0f);
  if (d == 0) flen[b] = ffalse;
}

// ---------------------------------------------------------------------------
// Block-tiled WMMA GEMM: C[M,N] = A[M,K] @ B[K,N] (+ bias[N]).
// Block: 256 threads / 8 waves; tile BM=128 x BN=64; BK=32.
// Requirements: M%16==0, N%64==0, K%32==0 (all call sites satisfy).
// B k-slab staged f32->f16 transposed into LDS [64 cols][32 k] (4 KB), so a
// lane's 16-element B fragment (col n, K=16h..16h+15) is 32 contiguous bytes.
// Fragment layouts used here (wave32):
//   A 16x32: lane L -> m=L%16, h=L/16; e -> K=16h+e.
//   B 32x16: lane L -> n=L%16; e -> K=16h+e.
//   C 16x16: vgpr r, lane L -> M=r+8h, N=L%16.
// ---------------------------------------------------------------------------
__global__ __launch_bounds__(256) void gemm_wmma(
    const float* __restrict__ A, const float* __restrict__ B,
    const float* __restrict__ bias, float* __restrict__ C,
    int M, int N, int K) {
  __shared__ _Float16 bt[64 * 32];            // [col][k], 4 KB

  int gn = N >> 6;                            // col blocks of 64
  int bmi = blockIdx.x / gn;
  int bni = blockIdx.x - bmi * gn;
  int n0 = bni * 64;

  int t = threadIdx.x;
  int w = t >> 5;                             // wave id 0..7
  int lane = t & 31;
  int h = lane >> 4, mn = lane & 15;

  int mbase0 = bmi * 128 + w * 16;            // this wave's m-tile
  bool valid = (mbase0 + 16 <= M);
  int mbase = valid ? mbase0 : (M - 16);      // clamp (dup compute, no store)

  // staging coords: thread handles col c, k rows kb..kb+7
  int c = t & 63;
  int kb = (t >> 6) * 8;

  const float* Arow = A + (size_t)(mbase + mn) * K;

  v8f acc0 = {}, acc1 = {}, acc2 = {}, acc3 = {};

  for (int k0 = 0; k0 < K; k0 += 32) {
    // ---- stage B slab (f32 -> f16, transposed) ----
    v8h pk;
#pragma unroll
    for (int i = 0; i < 8; ++i)
      pk[i] = (_Float16)B[(size_t)(k0 + kb + i) * N + n0 + c];
    *(v8h*)(bt + c * 32 + kb) = pk;           // 16B aligned ds_store
    __syncthreads();

    // ---- A fragment: one contiguous 16-float run -> 4x b128 loads ----
    const float4* ap4 = (const float4*)(Arow + k0 + 16 * h);
    float4 x0 = ap4[0], x1 = ap4[1];          // K = 16h+0..7
    float4 x2 = ap4[2], x3 = ap4[3];          // K = 16h+8..15
    v16h a;
    a[0] = (_Float16)x0.x; a[1] = (_Float16)x0.y; a[2] = (_Float16)x0.z; a[3] = (_Float16)x0.w;
    a[4] = (_Float16)x1.x; a[5] = (_Float16)x1.y; a[6] = (_Float16)x1.z; a[7] = (_Float16)x1.w;
    a[8]  = (_Float16)x2.x; a[9]  = (_Float16)x2.y; a[10] = (_Float16)x2.z; a[11] = (_Float16)x2.w;
    a[12] = (_Float16)x3.x; a[13] = (_Float16)x3.y; a[14] = (_Float16)x3.z; a[15] = (_Float16)x3.w;

    // ---- 4 n-tiles: B frags are 32 contiguous bytes in LDS ----
    v16h b0 = *(const v16h*)(bt + (0 * 16 + mn) * 32 + 16 * h);
    v16h b1 = *(const v16h*)(bt + (1 * 16 + mn) * 32 + 16 * h);
    v16h b2 = *(const v16h*)(bt + (2 * 16 + mn) * 32 + 16 * h);
    v16h b3 = *(const v16h*)(bt + (3 * 16 + mn) * 32 + 16 * h);
    acc0 = __builtin_amdgcn_wmma_f32_16x16x32_f16(false, a, false, b0, (short)0, acc0, false, false);
    acc1 = __builtin_amdgcn_wmma_f32_16x16x32_f16(false, a, false, b1, (short)0, acc1, false, false);
    acc2 = __builtin_amdgcn_wmma_f32_16x16x32_f16(false, a, false, b2, (short)0, acc2, false, false);
    acc3 = __builtin_amdgcn_wmma_f32_16x16x32_f16(false, a, false, b3, (short)0, acc3, false, false);
    __syncthreads();                          // protect bt before next stage
  }

  if (!valid) return;                         // wave-uniform; WMMAs already done
  v8f accs[4] = {acc0, acc1, acc2, acc3};
#pragma unroll
  for (int nt = 0; nt < 4; ++nt) {
    int n = n0 + nt * 16 + mn;
    float bb = bias ? bias[n] : 0.0f;         // hoisted: one load per column
#pragma unroll
    for (int r = 0; r < 8; ++r)
      C[(size_t)(mbase + r + 8 * h) * N + n] = accs[nt][r] + bb;
  }
}

// ---------------------------------------------------------------------------
// z = mean + eps*exp(0.5*lv); KL partial sums per block (deterministic tree).
// ---------------------------------------------------------------------------
__global__ __launch_bounds__(256) void latent_kernel(
    const float* __restrict__ mean, const float* __restrict__ lvar,
    const float* __restrict__ eps, float* __restrict__ z,
    float* __restrict__ klp) {
  __shared__ float sd[256];
  int i = blockIdx.x * 256 + threadIdx.x;     // i < B*L = 32768 exactly
  float mu = mean[i], lv = lvar[i];
  z[i] = mu + eps[i] * expf(0.5f * lv);
  sd[threadIdx.x] = 1.0f + lv - mu * mu - expf(lv);
  __syncthreads();
  for (int st = 128; st > 0; st >>= 1) {
    if (threadIdx.x < st) sd[threadIdx.x] += sd[threadIdx.x + st];
    __syncthreads();
  }
  if (threadIdx.x == 0) klp[blockIdx.x] = sd[0];
}

// ---------------------------------------------------------------------------
// Token loss: one wave per (b,j) position, 8 dims/lane; BOS/EOS/PAD sequence
// recomputed from ffalse. All gathers L2-resident.
// ---------------------------------------------------------------------------
__global__ __launch_bounds__(256) void loss_kernel(
    const int* __restrict__ vocab, const int* __restrict__ flen,
    const float* __restrict__ embW, const float* __restrict__ mW,
    const float* __restrict__ emb, float* __restrict__ lossp) {
  __shared__ float sd[256];
  int t = threadIdx.x;
  int p = blockIdx.x * 8 + (t >> 5);          // position < B*(S+1) = 131328
  int lane = t & 31;
  int b = p / (Ssz + 1), j = p - b * (Ssz + 1);
  int len = flen[b];
  int inp, tgt;
  if (j == 0)        inp = BOS_ID;
  else if (j <= len) inp = vocab[b * Ssz + j - 1];
  else               inp = PAD_ID;
  if (j < len)       tgt = vocab[b * Ssz + j];
  else if (j == len) tgt = (len == Ssz) ? BOS_ID : EOS_ID;  // roll wrap vs EOS
  else               tgt = PAD_ID;
  float acc = 0.0f;
  if (tgt != PAD_ID) {
    const float* wrow = embW + (size_t)inp * Dsz;
    const float* trow = emb  + (size_t)tgt * Dsz;
    const float* mrow = mW   + (size_t)b   * Dsz;
#pragma unroll
    for (int i = 0; i < 8; ++i) {
      int d = lane + 32 * i;
      float diff = wrow[d] + mrow[d] - trow[d];
      acc += diff * diff;
    }
  }
  sd[t] = acc;
  __syncthreads();
  for (int st = 128; st > 0; st >>= 1) {
    if (t < st) sd[t] += sd[t + st];
    __syncthreads();
  }
  if (t == 0) lossp[blockIdx.x] = sd[0];
}

// ---------------------------------------------------------------------------
// Final reduce: token_loss = sum(lossp)/max(valid*D,1);  kl = -0.5*sum/B.
// valid per batch = ffalse+1 (EOS / wrapped BOS target is always valid).
// ---------------------------------------------------------------------------
__global__ __launch_bounds__(256) void finalize_kernel(
    const float* __restrict__ lossp, int nloss,
    const float* __restrict__ klp, int nkl,
    const int* __restrict__ flen, float* __restrict__ out) {
  __shared__ float sd[256];
  int t = threadIdx.x;
  float s = 0.0f;
  for (int i = t; i < nloss; i += 256) s += lossp[i];
  sd[t] = s; __syncthreads();
  for (int st = 128; st > 0; st >>= 1) { if (t < st) sd[t] += sd[t + st]; __syncthreads(); }
  float losssum = sd[0]; __syncthreads();
  float k = 0.0f;
  for (int i = t; i < nkl; i += 256) k += klp[i];
  sd[t] = k; __syncthreads();
  for (int st = 128; st > 0; st >>= 1) { if (t < st) sd[t] += sd[t + st]; __syncthreads(); }
  float klsum = sd[0]; __syncthreads();
  sd[t] = (t < Bsz) ? (float)(flen[t] + 1) : 0.0f;
  __syncthreads();
  for (int st = 128; st > 0; st >>= 1) { if (t < st) sd[t] += sd[t + st]; __syncthreads(); }
  if (t == 0) {
    float denom = fmaxf(sd[0] * (float)Dsz, 1.0f);
    out[0] = losssum / denom;
    out[1] = -0.5f * klsum / (float)Bsz;
  }
}

extern "C" void kernel_launch(void* const* d_in, const int* in_sizes, int n_in,
                              void* d_out, int out_size, void* d_ws, size_t ws_size,
                              hipStream_t stream) {
  (void)in_sizes; (void)n_in; (void)out_size; (void)ws_size;
  const int*   vocab = (const int*)d_in[0];
  const int*   order = (const int*)d_in[1];
  const int*   mask  = (const int*)d_in[2];   // bool mask (integer per harness)
  const float* eps   = (const float*)d_in[3];
  const float* emb   = (const float*)d_in[4];
  const float* Wm    = (const float*)d_in[5];
  const float* bm    = (const float*)d_in[6];
  const float* Wv    = (const float*)d_in[7];
  const float* bv    = (const float*)d_in[8];
  const float* Wlin  = (const float*)d_in[9];
  const float* blin  = (const float*)d_in[10];
  const float* Wout  = (const float*)d_in[11];
  const float* bout  = (const float*)d_in[12];
  float* out = (float*)d_out;

  // Workspace layout (~53 MB)
  float* embW = (float*)d_ws;                       // V*D
  float* enc  = embW + (size_t)Vsz * Dsz;           // B*D
  float* mean = enc  + (size_t)Bsz * Dsz;           // B*L
  float* lvar = mean + (size_t)Bsz * Lsz;           // B*L
  float* z    = lvar + (size_t)Bsz * Lsz;           // B*L
  float* mem  = z    + (size_t)Bsz * Lsz;           // B*D
  float* mW   = mem  + (size_t)Bsz * Dsz;           // B*D
  int*   flen = (int*)(mW + (size_t)Bsz * Dsz);     // B
  float* klp  = (float*)(flen + Bsz);               // 128
  float* lossp = klp + (Bsz * Lsz / 256);           // 16416

  const int nklBlocks   = Bsz * Lsz / 256;                 // 128
  const int nLossBlocks = Bsz * (Ssz + 1) / 8;             // 16416

  // GEMM grids: gm = ceil(M/128), gn = N/64
  #define GEMM_GRID(M_, N_) ((((M_) + 127) / 128) * ((N_) / 64))

  // 1) pooled encoder + sequence lengths
  enc_kernel<<<Bsz, 256, 0, stream>>>(vocab, order, mask, emb, enc, flen);
  // 2) mean / log_var  (M=256, N=128, K=256)
  gemm_wmma<<<GEMM_GRID(Bsz, Lsz), 256, 0, stream>>>(enc, Wm, bm, mean, Bsz, Lsz, Dsz);
  gemm_wmma<<<GEMM_GRID(Bsz, Lsz), 256, 0, stream>>>(enc, Wv, bv, lvar, Bsz, Lsz, Dsz);
  // 3) reparameterize + KL partials
  latent_kernel<<<nklBlocks, 256, 0, stream>>>(mean, lvar, eps, z, klp);
  // 4) memory = z @ Wlin + blin  (M=256, N=256, K=128)
  gemm_wmma<<<GEMM_GRID(Bsz, Dsz), 256, 0, stream>>>(z, Wlin, blin, mem, Bsz, Dsz, Lsz);
  // 5) mW = memory @ Wout + bout (bout folded here, not into embW)
  gemm_wmma<<<GEMM_GRID(Bsz, Dsz), 256, 0, stream>>>(mem, Wout, bout, mW, Bsz, Dsz, Dsz);
  // 6) embW = emb @ Wout  (M=50000 -> 391*4 = 1564 blocks)
  gemm_wmma<<<GEMM_GRID(Vsz, Dsz), 256, 0, stream>>>(emb, Wout, nullptr, embW, Vsz, Dsz, Dsz);
  // 7) token-loss partials over all (b, j) positions
  loss_kernel<<<nLossBlocks, 256, 0, stream>>>(vocab, flen, embW, mW, emb, lossp);
  // 8) final scalars
  finalize_kernel<<<1, 256, 0, stream>>>(lossp, nLossBlocks, klp, nklBlocks, flen, out);
}
